// SelfAttention_22754736734347
// MI455X (gfx1250) — matrix operand order repaired and checked by reading, not compiled, not run on hardware
//
#include <hip/hip_runtime.h>

// ---------------------------------------------------------------------------
// Causal self-attention, B=4 T=4096 C=1024 HS=128, single head.
// Kernel 1: fused QKV projection (fp32 -> f16, v_wmma_f32_16x16x32_f16),
//           q pre-scaled by 1/sqrt(HS), k row-major, v transposed (HS,T).
// Kernel 2: flash attention (online softmax). K/V tiles staged into LDS by
//           the Tensor Data Mover (tensor_load_to_lds, TENSORcnt, double
//           buffered) and consumed by WMMA; P relayout through LDS.
// ---------------------------------------------------------------------------

typedef _Float16 half_t;
typedef __attribute__((ext_vector_type(16))) _Float16 v16h;
typedef __attribute__((ext_vector_type(8)))  _Float16 v8h;
typedef __attribute__((ext_vector_type(8)))  float    v8f;
typedef __attribute__((ext_vector_type(4)))  float    v4f;
typedef __attribute__((ext_vector_type(4)))  unsigned int v4u;
typedef __attribute__((ext_vector_type(8)))  int      v8i;
typedef __attribute__((ext_vector_type(4)))  int      v4i;

#define B_  4
#define T_  4096
#define C_  1024
#define HS_ 128

#if __has_builtin(__builtin_amdgcn_tensor_load_to_lds)
#define HAVE_TDM 1
#else
#define HAVE_TDM 0
#endif

union Frag16 { v16h v; v8h h[2]; };

static __device__ __forceinline__ v8f wmma_f16(const Frag16& a, const Frag16& b, v8f c) {
  // v_wmma_f32_16x16x32_f16: D = A(16x32) * B(32x16) + C(16x16 f32)
  return __builtin_amdgcn_wmma_f32_16x16x32_f16(false, a.v, false, b.v,
                                                (short)0, c, false, false);
}

#if HAVE_TDM
// Issue one 2D-tile TDM load (Tensor DMA Descriptor per CDNA5 ISA §8).
// tile is tile1 x tile0 elements (2 bytes each), row stride = stride0 elems.
// Tracked with TENSORcnt; EXEC is ignored by hardware.
static __device__ __forceinline__ void tdm_load_2d(
    unsigned int ldsAddr, unsigned long long gaddr,
    unsigned int tdim0, unsigned int tdim1,
    unsigned int tile0, unsigned int tile1,
    unsigned long long stride0)
{
  v4u g0;
  g0.x = 1u;                                   // count=1, user descriptor
  g0.y = ldsAddr;                              // lds_addr [63:32]
  g0.z = (unsigned int)gaddr;                  // global_addr [95:64]
  g0.w = (unsigned int)((gaddr >> 32) & 0x01FFFFFFull) | (2u << 30); // type=2
  v8i g1;
  g1[0] = (int)(1u << 16);                     // data_size=1 (2 bytes)
  g1[1] = (int)((tdim0 & 0xFFFFu) << 16);      // tensor_dim0[15:0] @ bit48
  g1[2] = (int)((tdim0 >> 16) | ((tdim1 & 0xFFFFu) << 16));
  g1[3] = (int)((tdim1 >> 16) | (tile0 << 16)); // tile_dim0 @ bit112
  g1[4] = (int)(tile1 & 0xFFFFu);              // tile_dim1 @ bit128
  g1[5] = (int)(unsigned int)(stride0 & 0xFFFFFFFFull); // dim0_stride @ bit160
  g1[6] = (int)(unsigned int)((stride0 >> 32) & 0xFFFFull);
  g1[7] = 0;
  v4i z4 = {0, 0, 0, 0};                       // groups 2/3 unused (2D tensor)
  v8i z8 = {0, 0, 0, 0, 0, 0, 0, 0};
  __builtin_amdgcn_tensor_load_to_lds(g0, g1, z4, z4, z8, 0);
}
#endif

// ---------------------------------------------------------------------------
// Kernel 1: QKV projection. grid = (B*T/128, 3), block = 256 (8 waves).
// ---------------------------------------------------------------------------
__global__ __launch_bounds__(256) void fa_qkv_proj(
    const float* __restrict__ x,
    const float* __restrict__ Wq, const float* __restrict__ Wk,
    const float* __restrict__ Wv,
    half_t* __restrict__ qh, half_t* __restrict__ kh, half_t* __restrict__ vt)
{
  constexpr int LDA = 40;                    // halfs per LDS row (80B, 16B-aligned)
  __shared__ half_t sA[128 * LDA];           // x tile, row-major [m][k]
  __shared__ half_t sB[128 * LDA];           // W tile, col-major [n][k]

  const int mat   = blockIdx.y;
  const float* W  = (mat == 0) ? Wq : ((mat == 1) ? Wk : Wv);
  const int mbase = blockIdx.x * 128;
  const int tid   = threadIdx.x;
  const int lane  = tid & 31;
  const int wave  = tid >> 5;

  v8f acc[8] = {};

  for (int ks = 0; ks < C_ / 32; ++ks) {
    const int k0 = ks * 32;
    __syncthreads();
    #pragma unroll
    for (int i = 0; i < 4; ++i) {
      const int idx = tid + i * 256;         // 128 rows x 8 float4-groups
      const int r   = idx >> 3;
      const int cg  = (idx & 7) * 4;
      v4f xv = *(const v4f*)(x + (size_t)(mbase + r) * C_ + k0 + cg);
      half_t* d = sA + r * LDA + cg;
      d[0] = (half_t)xv.x; d[1] = (half_t)xv.y;
      d[2] = (half_t)xv.z; d[3] = (half_t)xv.w;
    }
    #pragma unroll
    for (int i = 0; i < 4; ++i) {
      const int idx = tid + i * 256;         // 32 k-rows x 32 n-groups
      const int kk  = idx >> 5;
      const int ng  = (idx & 31) * 4;
      v4f wv = *(const v4f*)(W + (size_t)(k0 + kk) * HS_ + ng);
      sB[(ng + 0) * LDA + kk] = (half_t)wv.x;
      sB[(ng + 1) * LDA + kk] = (half_t)wv.y;
      sB[(ng + 2) * LDA + kk] = (half_t)wv.z;
      sB[(ng + 3) * LDA + kk] = (half_t)wv.w;
    }
    __syncthreads();

    const int arow = wave * 16 + (lane & 15);
    const int koff = (lane < 16) ? 0 : 16;   // A: K 0..15 / 16..31
    Frag16 a;
    a.h[0] = *(const v8h*)(sA + arow * LDA + koff);
    a.h[1] = *(const v8h*)(sA + arow * LDA + koff + 8);

    const int boff = (lane < 16) ? 0 : 16;   // B: K 0..15 / 16..31
    const int bn   = lane & 15;
    #pragma unroll
    for (int nt = 0; nt < 8; ++nt) {
      Frag16 bf;
      const half_t* bp = sB + (nt * 16 + bn) * LDA + boff;
      bf.h[0] = *(const v8h*)bp;
      bf.h[1] = *(const v8h*)(bp + 8);
      acc[nt] = wmma_f16(a, bf, acc[nt]);
    }
  }

  const int rbase = wave * 16 + ((lane < 16) ? 0 : 8);
  const int col0  = lane & 15;
  const float qscale = 0.08838834764831845f; // 1/sqrt(128)
  #pragma unroll
  for (int nt = 0; nt < 8; ++nt) {
    #pragma unroll
    for (int r = 0; r < 8; ++r) {
      const int grow = mbase + rbase + r;
      const int gcol = nt * 16 + col0;
      const float val = acc[nt][r];
      if (mat == 0) {
        qh[(size_t)grow * HS_ + gcol] = (half_t)(val * qscale);
      } else if (mat == 1) {
        kh[(size_t)grow * HS_ + gcol] = (half_t)val;
      } else {
        const int bb = grow >> 12;           // / T_
        const int tt = grow & (T_ - 1);
        vt[((size_t)bb * HS_ + gcol) * T_ + tt] = (half_t)val;  // (B,HS,T)
      }
    }
  }
}

// ---------------------------------------------------------------------------
// Kernel 2: flash attention. grid = (T/128, B), block = 256 (8 waves).
// K/V tiles double-buffered in LDS via TDM; wave w owns 16 query rows.
// ---------------------------------------------------------------------------
__global__ __launch_bounds__(256) void fa_attn(
    const half_t* __restrict__ qh, const half_t* __restrict__ kh,
    const half_t* __restrict__ vt, float* __restrict__ out)
{
  constexpr int LDP = 40;
  __shared__ half_t sK[2][32 * 128];         // [s][hs] per buffer (8 KB each)
  __shared__ half_t sV[2][128 * 32];         // [hs][s] per buffer (8 KB each)
  __shared__ half_t sP[8 * 16 * LDP];        // per-wave P relayout scratch

  const int bidx = blockIdx.y;
  const int tid  = threadIdx.x;
  const int lane = tid & 31;
  const int wave = tid >> 5;
  const int Rblk = blockIdx.x * 128;             // workgroup query-row base
  const int row0 = Rblk + wave * 16;             // this wave's query rows
  const size_t qkBase = (size_t)bidx * T_ * HS_;
  const size_t vtBase = (size_t)bidx * HS_ * T_;

  // q fragments (HS=128 -> 4 K-steps) held in registers for the whole loop
  const int qrow = row0 + (lane & 15);
  const int koff = (lane < 16) ? 0 : 16;
  Frag16 qf[4];
  #pragma unroll
  for (int kk = 0; kk < 4; ++kk) {
    const half_t* p = qh + qkBase + (size_t)qrow * HS_ + kk * 32 + koff;
    qf[kk].h[0] = *(const v8h*)p;
    qf[kk].h[1] = *(const v8h*)(p + 8);
  }

  v8f acc[8] = {};
  float mrow[8], lrow[8];
  #pragma unroll
  for (int r = 0; r < 8; ++r) { mrow[r] = -__builtin_inff(); lrow[r] = 0.0f; }

  const int rhalf   = (lane < 16) ? 0 : 8;   // C/D half-wave row offset
  const int colLane = lane & 15;
  const int boff    = (lane < 16) ? 0 : 16;  // B-fragment K offset

  const int nkbWave = (row0 + 15) / 32;      // last key block this wave needs
  const int nkbMax  = (Rblk + 127) / 32 + 1; // key blocks for the workgroup

  // --- stage key/value block into LDS buffer -------------------------------
#if HAVE_TDM
  // Wave 0 drives the Tensor Data Mover (2 descriptors per key block).
  #define STAGE_KV(bufi, s0v)                                                  \
    do { if (wave == 0) {                                                      \
      tdm_load_2d((unsigned int)(size_t)&sK[(bufi)][0],                        \
                  (unsigned long long)(size_t)(kh + qkBase + (size_t)(s0v) * HS_), \
                  HS_, 32, HS_, 32, HS_);                                      \
      tdm_load_2d((unsigned int)(size_t)&sV[(bufi)][0],                        \
                  (unsigned long long)(size_t)(vt + vtBase + (size_t)(s0v)),   \
                  32, HS_, 32, HS_, T_);                                       \
    } } while (0)
#else
  // Fallback: cooperative copy by all 256 threads (16 halfs each).
  #define STAGE_KV(bufi, s0v)                                                  \
    do {                                                                       \
      { const half_t* src = kh + qkBase + (size_t)(s0v) * HS_;  /* contiguous */\
        v8h a0 = *(const v8h*)(src + tid * 16);                                \
        v8h a1 = *(const v8h*)(src + tid * 16 + 8);                            \
        *(v8h*)(&sK[(bufi)][tid * 16])     = a0;                               \
        *(v8h*)(&sK[(bufi)][tid * 16 + 8]) = a1; }                             \
      { const int hsr = tid >> 1; const int of = (tid & 1) * 16;               \
        const half_t* src = vt + vtBase + (size_t)hsr * T_ + (s0v) + of;       \
        v8h a0 = *(const v8h*)src;                                             \
        v8h a1 = *(const v8h*)(src + 8);                                       \
        *(v8h*)(&sV[(bufi)][hsr * 32 + of])     = a0;                          \
        *(v8h*)(&sV[(bufi)][hsr * 32 + of + 8]) = a1; }                        \
    } while (0)
#endif

  STAGE_KV(0, 0);                            // prefetch first block

  for (int kb = 0; kb < nkbMax; ++kb) {
    const int s0  = kb * 32;
    const int cur = kb & 1;

    if (kb + 1 < nkbMax) STAGE_KV((kb + 1) & 1, s0 + 32);   // prefetch next
#if HAVE_TDM
    if (wave == 0) {
      // 2 TDM ops per block, in-order: <=2 outstanding means block kb landed.
      if (kb + 1 < nkbMax) __builtin_amdgcn_s_wait_tensorcnt((short)2);
      else                 __builtin_amdgcn_s_wait_tensorcnt((short)0);
    }
#endif
    __syncthreads();                         // buffer `cur` ready for all waves

    if (kb <= nkbWave) {                     // wave-uniform causal skip
      const half_t* kT = &sK[cur][0];
      const half_t* vT = &sV[cur][0];

      // S = q @ k^T (two 16x16 tiles covering 32 keys)
      v8f S0 = {}, S1 = {};
      #pragma unroll
      for (int kk = 0; kk < 4; ++kk) {
        Frag16 b0, b1;
        const half_t* p0 = kT + (size_t)colLane * HS_ + kk * 32 + boff;
        b0.h[0] = *(const v8h*)p0;  b0.h[1] = *(const v8h*)(p0 + 8);
        const half_t* p1 = kT + (size_t)(16 + colLane) * HS_ + kk * 32 + boff;
        b1.h[0] = *(const v8h*)p1;  b1.h[1] = *(const v8h*)(p1 + 8);
        S0 = wmma_f16(qf[kk], b0, S0);
        S1 = wmma_f16(qf[kk], b1, S1);
      }

      // causal mask (branchless; EXEC stays all-ones)
      #pragma unroll
      for (int r = 0; r < 8; ++r) {
        const int grow = row0 + rhalf + r;
        S0[r] = (s0 + colLane      > grow) ? -__builtin_inff() : S0[r];
        S1[r] = (s0 + 16 + colLane > grow) ? -__builtin_inff() : S1[r];
      }

      // online softmax: row reductions over 16 lanes (stay inside half-wave)
      float alpha[8];
      #pragma unroll
      for (int r = 0; r < 8; ++r) {
        float mx = fmaxf(S0[r], S1[r]);
        #pragma unroll
        for (int m = 8; m >= 1; m >>= 1) mx = fmaxf(mx, __shfl_xor(mx, m, 32));
        const float mnew = fmaxf(mrow[r], mx);
        alpha[r] = __expf(mrow[r] - mnew);
        mrow[r]  = mnew;
        const float p0 = __expf(S0[r] - mnew);
        const float p1 = __expf(S1[r] - mnew);
        S0[r] = p0; S1[r] = p1;
        float rs = p0 + p1;
        #pragma unroll
        for (int m = 8; m >= 1; m >>= 1) rs += __shfl_xor(rs, m, 32);
        lrow[r] = lrow[r] * alpha[r] + rs;
      }
      #pragma unroll
      for (int nt = 0; nt < 8; ++nt)
        #pragma unroll
        for (int r = 0; r < 8; ++r) acc[nt][r] *= alpha[r];

      // P: C/D layout -> A-fragment layout via per-wave LDS scratch
      half_t* wp = sP + wave * 16 * LDP;
      #pragma unroll
      for (int r = 0; r < 8; ++r) {
        const int rl = rhalf + r;
        wp[rl * LDP + colLane]      = (half_t)S0[r];
        wp[rl * LDP + 16 + colLane] = (half_t)S1[r];
      }
      asm volatile("" ::: "memory");         // same-wave LDS ops stay in order
      Frag16 pf;
      {
        const half_t* p = wp + (lane & 15) * LDP + koff;
        pf.h[0] = *(const v8h*)p;
        pf.h[1] = *(const v8h*)(p + 8);
      }

      // O += P @ V
      #pragma unroll
      for (int nt = 0; nt < 8; ++nt) {
        Frag16 bv;
        const half_t* vp = vT + (size_t)(nt * 16 + colLane) * 32 + boff;
        bv.h[0] = *(const v8h*)vp;
        bv.h[1] = *(const v8h*)(vp + 8);
        acc[nt] = wmma_f16(pf, bv, acc[nt]);
      }
    }
    __syncthreads();                         // all reads of `cur` done before
  }                                          // it gets rewritten by the TDM

  // epilogue: normalize and store fp32
  #pragma unroll
  for (int nt = 0; nt < 8; ++nt) {
    #pragma unroll
    for (int r = 0; r < 8; ++r) {
      const int grow = row0 + rhalf + r;
      const int gcol = nt * 16 + colLane;
      out[((size_t)bidx * T_ + grow) * HS_ + gcol] = acc[nt][r] / lrow[r];
    }
  }
}

// ---------------------------------------------------------------------------
extern "C" void kernel_launch(void* const* d_in, const int* in_sizes, int n_in,
                              void* d_out, int out_size, void* d_ws, size_t ws_size,
                              hipStream_t stream) {
  (void)in_sizes; (void)n_in; (void)out_size; (void)ws_size;
  const float* x  = (const float*)d_in[0];
  const float* Wq = (const float*)d_in[1];
  const float* Wk = (const float*)d_in[2];
  const float* Wv = (const float*)d_in[3];
  float* out = (float*)d_out;

  // workspace: q (scaled, f16), k (f16, (B*T,HS)), v^T (f16, (B,HS,T)) = 12 MB
  half_t* qh = (half_t*)d_ws;
  half_t* kh = qh + (size_t)B_ * T_ * HS_;
  half_t* vt = kh + (size_t)B_ * T_ * HS_;

  dim3 g1(B_ * T_ / 128, 3);
  fa_qkv_proj<<<g1, 256, 0, stream>>>(x, Wq, Wk, Wv, qh, kh, vt);

  dim3 g2(T_ / 128, B_);
  fa_attn<<<g2, 256, 0, stream>>>(qh, kh, vt, out);
}
